// HUGOSAttention_50027779064481
// MI455X (gfx1250) — compile-verified, matrix operand order checked by reading, not run on hardware
//
#include <hip/hip_runtime.h>
#include <hip/hip_bf16.h>
#include <math.h>

// Problem constants (from reference): B=1024, N=49, E=2048, D=512, A=512
constexpr int kB = 1024;
constexpr int kN = 49;
constexpr int kNp = 64;    // N padded to 4 WMMA row-tiles
constexpr int kE = 2048;
constexpr int kD = 512;
constexpr int kA = 512;

typedef __attribute__((ext_vector_type(16))) __bf16 v16bf;
typedef __attribute__((ext_vector_type(8)))  float  v8f;

union FragAB { v16bf v; unsigned u[8]; };

// fp32 -> bf16 (round to nearest even-ish)
__device__ __forceinline__ unsigned short f32_to_bf16(float x) {
    unsigned u = __float_as_uint(x);
    u += 0x7FFFu + ((u >> 16) & 1u);
    return (unsigned short)(u >> 16);
}
__device__ __forceinline__ float bf16_to_f32(unsigned short h) {
    return __uint_as_float(((unsigned)h) << 16);
}
__device__ __forceinline__ unsigned pack_bf16x2(float lo, float hi) {
    return ((unsigned)f32_to_bf16(hi) << 16) | (unsigned)f32_to_bf16(lo);
}

// Per-lane bf16 WMMA fragment k-offset (ISA 7.12.2, 16-bit A/B 16x32 layout)
__device__ __forceinline__ int frag_koff(int v, int hi) {
    return (v < 4) ? (hi * 8 + 2 * v) : (16 + hi * 8 + 2 * (v - 4));
}

// ---------------------------------------------------------------------------
// Prep: W_enc [A,E] fp32 -> bf16 in d_ws, pre-swizzled into WMMA B-fragment
// order. Layout: per (a-tile at, k-chunk c) a 1024-byte wave block:
//   [ 32 lanes x 16B : dwords v0..3 ][ 32 lanes x 16B : dwords v4..7 ]
// so the B-fragment load is two coalesced global_load_b128 per lane.
// ---------------------------------------------------------------------------
__global__ void wenc_swizzle_kernel(const float* __restrict__ w, unsigned* __restrict__ out) {
    const int idx = blockIdx.x * blockDim.x + threadIdx.x;  // output dword index
    if (idx >= kA * kE / 2) return;
    const int block = idx >> 8;            // 256 dwords per (at,c) block
    const int w8    = idx & 255;
    const int sel   = w8 >> 7;             // 0: v0..3, 1: v4..7
    const int w7    = w8 & 127;
    const int L     = w7 >> 2;             // lane
    const int v     = (w7 & 3) + sel * 4;  // dword index in fragment
    const int at    = block >> 6;
    const int c     = block & 63;
    const int a     = at * 16 + (L & 15);
    const int hi    = L >> 4;
    const int k     = c * 32 + frag_koff(v, hi);
    const float* src = w + (size_t)a * kE + k;
    out[idx] = pack_bf16x2(src[0], src[1]);
}

// ---------------------------------------------------------------------------
// Fused attention: one workgroup per batch element b.
// ---------------------------------------------------------------------------
// dyn-LDS layout (bytes). enc tile stored fragment-swizzled:
// per (m-tile, k-chunk c): 1024B wave block as above; block index = m*64 + c.
constexpr int SM_ENC   = 0;                         // 4*64*1024 = 262144
constexpr int SM_ATT2  = SM_ENC + kNp * kE * 2;     // float[512]
constexpr int SM_WFULL = SM_ATT2 + kA * 4;          // float[512]
constexpr int SM_DH    = SM_WFULL + kA * 4;         // float[512]
constexpr int SM_ATT   = SM_DH + kD * 4;            // float[64]
constexpr int SM_ALPHA = SM_ATT + kNp * 4;          // float[64]
constexpr int SM_SCAL  = SM_ALPHA + kNp * 4;        // float[8]
constexpr int SMEM_BYTES = SM_SCAL + 8 * 4;         // = 268832

__global__ __launch_bounds__(256, 1)
void fused_attn_kernel(const float* __restrict__ enc,      // [B,N,E]
                       const float* __restrict__ dh,       // [B,D]
                       const unsigned* __restrict__ wbf,   // W_enc bf16, swizzled
                       const float* __restrict__ b_enc,    // [A]
                       const float* __restrict__ W_dec,    // [A,D]
                       const float* __restrict__ b_dec,    // [A]
                       const float* __restrict__ W_sig,    // [1,D]
                       const float* __restrict__ b_sig,    // [1]
                       const float* __restrict__ W_full,   // [1,A]
                       float* __restrict__ out_awe,        // [B,E]
                       float* __restrict__ out_alpha,      // [B,N]
                       float* __restrict__ out_beta)       // [B,1]
{
    extern __shared__ char smem[];
    char*  enc_s   = smem + SM_ENC;
    float* att2_s  = (float*)(smem + SM_ATT2);
    float* wfull_s = (float*)(smem + SM_WFULL);
    float* dh_s    = (float*)(smem + SM_DH);
    float* att_s   = (float*)(smem + SM_ATT);
    float* alpha_s = (float*)(smem + SM_ALPHA);
    float* scal    = (float*)(smem + SM_SCAL);

    const int b    = blockIdx.x;
    const int tid  = threadIdx.x;
    const int wave = tid >> 5;
    const int lane = tid & 31;
    const int lr   = lane & 15;     // lane within half-wave
    const int hi   = lane >> 4;     // half-wave select

    // ---------------- Phase 1: stage operands ----------------
    for (int d = tid; d < kD; d += 256) dh_s[d] = dh[(size_t)b * kD + d];
    __syncthreads();

    // encoder_out[b] -> LDS as bf16 in fragment-swizzled order; rows>=49 zero.
    {
        const float4* ev = (const float4*)(enc + (size_t)b * kN * kE);
        for (int q = tid; q < kNp * (kE / 4); q += 256) {
            const int row = q >> 9;            // / (kE/4)
            const int c4  = q & 511;
            float4 vv;
            if (row < kN) vv = ev[row * (kE / 4) + c4];
            else          vv = make_float4(0.f, 0.f, 0.f, 0.f);
            uint2 p;
            p.x = pack_bf16x2(vv.x, vv.y);
            p.y = pack_bf16x2(vv.z, vv.w);
            // swizzled store address
            const int e   = c4 << 2;
            const int c   = e >> 5;
            const int r   = e & 31;
            const int sel = r >> 4;
            const int rr  = r & 15;
            const int h2  = rr >> 3;
            const int dwo = (rr & 7) >> 1;     // 0 or 2
            const int m   = row >> 4;
            const int L   = h2 * 16 + (row & 15);
            *(uint2*)(enc_s + (m * 64 + c) * 1024 + sel * 512 + L * 16 + dwo * 4) = p;
        }
    }

    // att2[a] = b_enc[a] + b_dec[a] + dot(dh, W_dec[a])
    for (int a = tid; a < kA; a += 256) {
        const float4* wr = (const float4*)(W_dec + (size_t)a * kD);
        float acc = b_enc[a] + b_dec[a];
        #pragma unroll 4
        for (int d4 = 0; d4 < kD / 4; ++d4) {
            float4 w4 = wr[d4];
            const float4 h4 = ((const float4*)dh_s)[d4];
            acc += w4.x * h4.x + w4.y * h4.y + w4.z * h4.z + w4.w * h4.w;
        }
        att2_s[a] = acc;
    }
    for (int a = tid; a < kA; a += 256) wfull_s[a] = W_full[a];
    if (tid < kNp) att_s[tid] = 0.f;

    // beta = sigmoid(dot(dh, W_sig) + b_sig) : wave 0
    if (tid < 32) {
        float p = 0.f;
        for (int d = lane; d < kD; d += 32) p += dh_s[d] * W_sig[d];
        #pragma unroll
        for (int off = 16; off > 0; off >>= 1) p += __shfl_xor(p, off, 32);
        if (lane == 0) {
            float beta = 1.f / (1.f + __expf(-(p + b_sig[0])));
            scal[2] = beta;
            out_beta[b] = beta;
        }
    }
    __syncthreads();

    // ---------------- Phase 2: WMMA GEMM + fused relu-dot reduction --------
    // 32 A-col tiles round-robined over 8 waves, 4 M-tiles each, K = 2048/32.
    for (int at = wave; at < kA / 16; at += 8) {
        const int a0 = at * 16;
        v8f acc[4];
        #pragma unroll
        for (int m = 0; m < 4; ++m) acc[m] = v8f{};

        for (int k0 = 0; k0 < kE; k0 += 32) {
            const int c = k0 >> 5;
            // B fragment: two coalesced 16B global loads (wave = 512B run)
            const unsigned* bp = wbf + ((size_t)(at * 64 + c) << 8) + lane * 4;
            __builtin_prefetch((const void*)(bp + 256), 0, 0);  // next k-chunk
            FragAB bf;
            *(uint4*)&bf.u[0] = *(const uint4*)bp;
            *(uint4*)&bf.u[4] = *(const uint4*)(bp + 128);

            #pragma unroll
            for (int m = 0; m < 4; ++m) {
                // A fragment: two conflict-free ds_load_b128
                const char* ap = enc_s + (m * 64 + c) * 1024 + lane * 16;
                FragAB af;
                *(uint4*)&af.u[0] = *(const uint4*)ap;
                *(uint4*)&af.u[4] = *(const uint4*)(ap + 512);
                acc[m] = __builtin_amdgcn_wmma_f32_16x16x32_bf16(
                    false, af.v, false, bf.v, (short)0, acc[m], false, false);
            }
        }

        // fused epilogue: relu(att1 + att2) * W_full, reduce over A into att_s
        const int a = a0 + lr;
        const float t2 = att2_s[a];
        const float wf = wfull_s[a];
        #pragma unroll
        for (int m = 0; m < 4; ++m) {
            #pragma unroll
            for (int v = 0; v < 8; ++v) {
                float val = acc[m][v] + t2;
                val = fmaxf(val, 0.f) * wf;
                #pragma unroll
                for (int off = 1; off < 16; off <<= 1)
                    val += __shfl_xor(val, off, 32);
                if (lr == 0)
                    atomicAdd(&att_s[m * 16 + hi * 8 + v], val);
            }
        }
    }
    __syncthreads();

    // ---------------- Phase 3: softmax over N=49, scale by beta -----------
    // (b_full is constant over n -> softmax-invariant; att itself not output)
    if (tid == 0) {
        float m = -INFINITY;
        for (int i = 0; i < kN; ++i) m = fmaxf(m, att_s[i]);
        scal[0] = m;
    }
    __syncthreads();
    if (tid < kN) alpha_s[tid] = __expf(att_s[tid] - scal[0]);
    __syncthreads();
    if (tid == 0) {
        float s = 0.f;
        for (int i = 0; i < kN; ++i) s += alpha_s[i];
        scal[1] = 1.f / s;
    }
    __syncthreads();
    if (tid < kN) {
        float a = alpha_s[tid] * scal[1] * scal[2];   // softmax * beta
        alpha_s[tid] = a;
        out_alpha[(size_t)b * kN + tid] = a;
    }
    __syncthreads();

    // ---------------- Phase 4: awe[b,e] = sum_n alpha[n]*enc[n,e] ----------
    // inverse swizzle: per e all terms fixed except (m, lr)
    for (int e = tid; e < kE; e += 256) {
        const int c   = e >> 5;
        const int r   = e & 31;
        const int sel = r >> 4;
        const int rr  = r & 15;
        const int h2  = rr >> 3;
        const int dwo = (rr & 7) >> 1;
        const int off0 = c * 1024 + sel * 512 + h2 * 256 + dwo * 4 + (e & 1) * 2;
        float acc = 0.f;
        #pragma unroll 7
        for (int n = 0; n < kN; ++n) {
            const int m   = n >> 4;
            const int lr2 = n & 15;
            acc += alpha_s[n] *
                   bf16_to_f32(*(const unsigned short*)(enc_s + off0 + m * 65536 + lr2 * 16));
        }
        out_awe[(size_t)b * kE + e] = acc;
    }
}

// ---------------------------------------------------------------------------
extern "C" void kernel_launch(void* const* d_in, const int* in_sizes, int n_in,
                              void* d_out, int out_size, void* d_ws, size_t ws_size,
                              hipStream_t stream) {
    const float* enc    = (const float*)d_in[0];   // [B,N,E]
    const float* dh     = (const float*)d_in[1];   // [B,D]
    const float* W_enc  = (const float*)d_in[2];   // [A,E]
    const float* b_enc  = (const float*)d_in[3];   // [A]
    const float* W_dec  = (const float*)d_in[4];   // [A,D]
    const float* b_dec  = (const float*)d_in[5];   // [A]
    const float* W_sig  = (const float*)d_in[6];   // [1,D]
    const float* b_sig  = (const float*)d_in[7];   // [1]
    const float* W_full = (const float*)d_in[8];   // [1,A]
    // d_in[9] = b_full : softmax shift-invariant, unused

    float* out       = (float*)d_out;
    float* out_awe   = out;                                   // [B,E]
    float* out_alpha = out + (size_t)kB * kE;                 // [B,N]
    float* out_beta  = out_alpha + (size_t)kB * kN;           // [B,1]

    unsigned* wbf_sw = (unsigned*)d_ws;                       // swizzled W_enc bf16 (2 MB)

    // Prep: fp32 -> bf16 W_enc, fragment-swizzled
    {
        const int ndw = kA * kE / 2;
        wenc_swizzle_kernel<<<(ndw + 255) / 256, 256, 0, stream>>>(W_enc, wbf_sw);
    }

    // Fused attention: 1 block / batch element, 256 threads (8 waves), ~262.5 KB LDS
    (void)hipFuncSetAttribute((const void*)fused_attn_kernel,
                              hipFuncAttributeMaxDynamicSharedMemorySize, SMEM_BYTES);
    fused_attn_kernel<<<kB, 256, SMEM_BYTES, stream>>>(
        enc, dh, wbf_sw, b_enc, W_dec, b_dec,
        W_sig, b_sig, W_full, out_awe, out_alpha, out_beta);
}
